// Encoder_76304388981261
// MI455X (gfx1250) — matrix operand order repaired and checked by reading, not compiled, not run on hardware
//
#include <hip/hip_runtime.h>

// ---------------------------------------------------------------------------
// Constants from the reference model
// ---------------------------------------------------------------------------
#define N_NODES   50000
#define N_HEDGES  10000
#define E_BASE    150000
#define E_TOT     200000   // E_BASE + N_NODES self edges
#define N_TGT     60000    // N_HEDGES + N_NODES
#define HID       768
#define HEADS     12
#define HDIM      64
#define LAYERS    4
#define TOKLEN    8
#define LN_EPS    1e-12f
#define KT        64       // GEMM LDS K-tile (2 WMMA k-steps per barrier)

// ---------------------------------------------------------------------------
// Vector types for WMMA (CDNA5 / gfx1250, wave32)
// ---------------------------------------------------------------------------
typedef __attribute__((ext_vector_type(8)))  float          f32x8;
typedef __attribute__((ext_vector_type(16))) __bf16         bf16x16;
typedef __attribute__((ext_vector_type(8)))  __bf16         bf16x8;
typedef __attribute__((ext_vector_type(8)))  unsigned short us8;
typedef __attribute__((ext_vector_type(4)))  unsigned short us4;

__device__ __forceinline__ f32x8 f32x8_zero() {
  f32x8 v;
#pragma unroll
  for (int i = 0; i < 8; ++i) v[i] = 0.0f;
  return v;
}

// float -> bf16 (round to nearest even)
__device__ __forceinline__ unsigned short f2bf(float f) {
  unsigned int u = __float_as_uint(f);
  u += 0x7FFFu + ((u >> 16) & 1u);
  return (unsigned short)(u >> 16);
}

// order-preserving float<->uint encoding for atomic max
__device__ __forceinline__ unsigned int enc_ord(float f) {
  unsigned int u = __float_as_uint(f);
  return (u & 0x80000000u) ? ~u : (u | 0x80000000u);
}
__device__ __forceinline__ float dec_ord(unsigned int u) {
  u = (u & 0x80000000u) ? (u ^ 0x80000000u) : ~u;
  return __uint_as_float(u);
}

// block-wide sum over 256 threads
__device__ __forceinline__ float block_sum(float v, float* s) {
  int t = threadIdx.x;
  s[t] = v;
  __syncthreads();
  for (int off = 128; off > 0; off >>= 1) {
    if (t < off) s[t] += s[t + off];
    __syncthreads();
  }
  float r = s[0];
  __syncthreads();
  return r;
}

// ---------------------------------------------------------------------------
// GEMM: C[M,N] = act(A_bf16[M,K] @ B_bf16[K,N] + bias), f32 accumulate via WMMA
// block = 256 threads (8 wave32), block tile 128x128, wave tile 32x64.
// A tile staged with GLOBAL_LOAD_ASYNC_TO_LDS_B128 (ASYNCcnt); B tile staged
// through VGPRs with a packed 4-wide transpose (b64 LDS stores).
// 16 WMMAs per barrier.
// ---------------------------------------------------------------------------
__global__ __launch_bounds__(256) void gemm_bf16_wmma(
    const unsigned short* __restrict__ A, const unsigned short* __restrict__ B,
    const float* __restrict__ bias, float* __restrict__ C,
    int M, int N, int K, int relu) {
  __shared__ __align__(32) unsigned short lA[2][128 * KT];
  __shared__ __align__(32) unsigned short lB[2][128 * KT];  // transposed [n][k]

  const int tid  = threadIdx.x;
  const int lane = tid & 31;
  const int wid  = tid >> 5;
  const int m0   = blockIdx.y * 128;
  const int n0   = blockIdx.x * 128;
  const int wm   = (wid >> 1) * 32;  // 4 wave rows
  const int wn   = (wid & 1) * 64;   // 2 wave cols

  f32x8 acc[2][4];
#pragma unroll
  for (int mi = 0; mi < 2; ++mi)
#pragma unroll
    for (int ni = 0; ni < 4; ++ni) acc[mi][ni] = f32x8_zero();

  const int nk = K / KT;

  auto load_stage = [&](int s, int k0) {
    {  // A tile: 128 rows x 64 cols -> async DMA straight into LDS
      int r = tid >> 1;
      int half = tid & 1;
      int grow = m0 + r;
      if (grow >= M) grow = M - 1;  // clamp: rows >= M are discarded at epilogue
      unsigned long long ga =
          (unsigned long long)(size_t)(A + (size_t)grow * K + k0 + half * 32);
      unsigned lds = (unsigned)(size_t)(&lA[s][r * KT + half * 32]);
      asm volatile(
          "global_load_async_to_lds_b128 %0, %1, off\n\t"
          "global_load_async_to_lds_b128 %0, %1, off offset:16\n\t"
          "global_load_async_to_lds_b128 %0, %1, off offset:32\n\t"
          "global_load_async_to_lds_b128 %0, %1, off offset:48"
          :: "v"(lds), "v"(ga) : "memory");
    }
    {  // B tile: 64 k-rows x 128 n-cols, packed transpose (b64 LDS stores)
      int kg = (tid >> 4) * 4;   // k-row group base 0,4,...,60
      int nc = (tid & 15) * 8;   // n-col base 0,8,...,120
      const unsigned short* bp = B + (size_t)(k0 + kg) * N + n0 + nc;
      us8 r0 = *(const us8*)(bp);
      us8 r1 = *(const us8*)(bp + N);
      us8 r2 = *(const us8*)(bp + 2 * N);
      us8 r3 = *(const us8*)(bp + 3 * N);
#pragma unroll
      for (int j = 0; j < 8; ++j) {
        us4 w;
        w[0] = r0[j]; w[1] = r1[j]; w[2] = r2[j]; w[3] = r3[j];
        *(us4*)(&lB[s][(nc + j) * KT + kg]) = w;
      }
    }
  };

  load_stage(0, 0);
  asm volatile("s_wait_asynccnt 0" ::: "memory");
  __syncthreads();

  for (int kt = 0; kt < nk; ++kt) {
    const int cur = kt & 1;
    if (kt + 2 < nk) {  // prefetch A two tiles ahead -> global_prefetch_b8
      int grow = m0 + (tid >> 1);
      if (grow < M) __builtin_prefetch(A + (size_t)grow * K + (kt + 2) * KT, 0, 1);
    }
    if (kt + 1 < nk) load_stage(cur ^ 1, (kt + 1) * KT);

    // Fragments: contiguous k order (k = 16*(lane/16) + s), same for A and B.
    const int rsel = lane & 15;
    const int klo  = (lane < 16) ? 0 : 16;  // A: lanes 0-15 K0-15; lanes 16-31 K16-31
    const int koff = (lane < 16) ? 0 : 16;  // B: lanes 0-15 K0-15; lanes 16-31 K16-31

#pragma unroll
    for (int kk = 0; kk < 2; ++kk) {
      bf16x16 afrag[2];
#pragma unroll
      for (int mi = 0; mi < 2; ++mi) {
        int row = wm + mi * 16 + rsel;
        union { bf16x16 v; bf16x8 h[2]; } u;
        u.h[0] = *(const bf16x8*)(&lA[cur][row * KT + kk * 32 + klo]);
        u.h[1] = *(const bf16x8*)(&lA[cur][row * KT + kk * 32 + klo + 8]);
        afrag[mi] = u.v;
      }
      bf16x16 bfrag[4];
#pragma unroll
      for (int ni = 0; ni < 4; ++ni) {
        int col = wn + ni * 16 + rsel;
        bfrag[ni] = *(const bf16x16*)(&lB[cur][col * KT + kk * 32 + koff]);
      }
#pragma unroll
      for (int mi = 0; mi < 2; ++mi)
#pragma unroll
        for (int ni = 0; ni < 4; ++ni)
          acc[mi][ni] = __builtin_amdgcn_wmma_f32_16x16x32_bf16(
              false, afrag[mi], false, bfrag[ni], (short)0, acc[mi][ni], false, false);
    }

    asm volatile("s_wait_asynccnt 0" ::: "memory");
    __syncthreads();
  }

  // Epilogue: 16x16 f32 C layout (VGPR v: M=v+8*(lane>=16), N=lane&15)
#pragma unroll
  for (int mi = 0; mi < 2; ++mi) {
#pragma unroll
    for (int ni = 0; ni < 4; ++ni) {
      int col = n0 + wn + ni * 16 + (lane & 15);
      float bv = bias ? bias[col] : 0.0f;
#pragma unroll
      for (int v = 0; v < 8; ++v) {
        int row = m0 + wm + mi * 16 + v + ((lane >> 4) << 3);
        if (row < M) {
          float x = acc[mi][ni][v] + bv;
          if (relu) x = fmaxf(x, 0.0f);
          C[(size_t)row * N + col] = x;
        }
      }
    }
  }
}

// ---------------------------------------------------------------------------
// f32 -> bf16 with strided/offset output (for concat packing)
// ---------------------------------------------------------------------------
__global__ void cvt_kernel(const float* __restrict__ in, unsigned short* __restrict__ out,
                           long long rows, int cols, int ldout, int coloff) {
  long long i = (long long)blockIdx.x * blockDim.x + threadIdx.x;
  if (i >= rows * (long long)cols) return;
  long long r = i / cols;
  int c = (int)(i - r * cols);
  out[r * (long long)ldout + coloff + c] = f2bf(in[i]);
}

__global__ void fill_u32_kernel(unsigned int* __restrict__ p, unsigned int v, long long n) {
  long long i = (long long)blockIdx.x * blockDim.x + threadIdx.x;
  if (i < n) p[i] = v;
}

// ---------------------------------------------------------------------------
// Token embedding mean-pool + LayerNorm. One block (256 thr) per row.
// table row 0 is zero so summing all tokens == summing nonzero ones.
// ---------------------------------------------------------------------------
__global__ void embed_pool_ln_kernel(const int* __restrict__ toks, const float* __restrict__ table,
                                     const float* __restrict__ g, const float* __restrict__ b,
                                     float* __restrict__ out) {
  const int row = blockIdx.x;
  __shared__ float red[256];
  __shared__ int ids[TOKLEN];
  const int t = threadIdx.x;
  if (t < TOKLEN) ids[t] = toks[row * TOKLEN + t];
  __syncthreads();
  int cnt = 0;
#pragma unroll
  for (int k = 0; k < TOKLEN; ++k) cnt += (ids[k] != 0);
  const float inv = 1.0f / (float)(cnt ? cnt : 1);

  float v[3];
#pragma unroll
  for (int i = 0; i < 3; ++i) {
    int c = t + i * 256;
    float s = 0.0f;
#pragma unroll
    for (int k = 0; k < TOKLEN; ++k) s += table[(size_t)ids[k] * HID + c];
    v[i] = s * inv;
  }
  float mean = block_sum(v[0] + v[1] + v[2], red) * (1.0f / HID);
  float d0 = v[0] - mean, d1 = v[1] - mean, d2 = v[2] - mean;
  float var = block_sum(d0 * d0 + d1 * d1 + d2 * d2, red) * (1.0f / HID);
  float rstd = rsqrtf(var + LN_EPS);
#pragma unroll
  for (int i = 0; i < 3; ++i) {
    int c = t + i * 256;
    out[(size_t)row * HID + c] = (v[i] - mean) * rstd * g[c] + b[c];
  }
}

// ---------------------------------------------------------------------------
// alpha[i,h] = sum_c xK[i,h,c] * att_r[h,c]
// ---------------------------------------------------------------------------
__global__ void alpha_kernel(const float* __restrict__ xK, const float* __restrict__ att_r,
                             float* __restrict__ alpha, int n) {
  long long i = (long long)blockIdx.x * blockDim.x + threadIdx.x;
  if (i >= (long long)n * HEADS) return;
  int row = (int)(i / HEADS), h = (int)(i % HEADS);
  const float* xk = xK + (size_t)row * HID + h * HDIM;
  const float* ar = att_r + h * HDIM;
  float s = 0.0f;
#pragma unroll 8
  for (int c = 0; c < HDIM; ++c) s += xk[c] * ar[c];
  alpha[i] = s;
}

// ---------------------------------------------------------------------------
// Segment softmax over edges (3 phases, f32 atomics through L2)
// ---------------------------------------------------------------------------
__global__ void seg_max_kernel(const float* __restrict__ alpha, const int* __restrict__ src,
                               const int* __restrict__ dst, unsigned int* __restrict__ amax) {
  long long i = (long long)blockIdx.x * blockDim.x + threadIdx.x;
  if (i >= (long long)E_TOT * HEADS) return;
  int e = (int)(i / HEADS), h = (int)(i % HEADS);
  float a = alpha[(size_t)src[e] * HEADS + h];
  a = (a > 0.0f) ? a : a * 0.2f;  // leaky_relu
  atomicMax(&amax[(size_t)dst[e] * HEADS + h], enc_ord(a));
}

__global__ void seg_exp_kernel(const float* __restrict__ alpha, const int* __restrict__ src,
                               const int* __restrict__ dst, const unsigned int* __restrict__ amax,
                               float* __restrict__ denom, float* __restrict__ ex) {
  long long i = (long long)blockIdx.x * blockDim.x + threadIdx.x;
  if (i >= (long long)E_TOT * HEADS) return;
  int e = (int)(i / HEADS), h = (int)(i % HEADS);
  float a = alpha[(size_t)src[e] * HEADS + h];
  a = (a > 0.0f) ? a : a * 0.2f;
  float m = dec_ord(amax[(size_t)dst[e] * HEADS + h]);
  if (!(fabsf(m) < 3.4e38f)) m = 0.0f;  // replicate isfinite() guard
  float v = __expf(a - m);
  ex[i] = v;
  atomicAdd(&denom[(size_t)dst[e] * HEADS + h], v);
}

__global__ void seg_msg_kernel(const float* __restrict__ ex, const float* __restrict__ denom,
                               const float* __restrict__ xV, const int* __restrict__ src,
                               const int* __restrict__ dst, float* __restrict__ out) {
  long long i = (long long)blockIdx.x * blockDim.x + threadIdx.x;
  if (i >= (long long)E_TOT * HEADS) return;
  int e = (int)(i / HEADS), h = (int)(i % HEADS);
  float w = ex[i] / (denom[(size_t)dst[e] * HEADS + h] + 1e-16f);
  const float* v = xV + (size_t)src[e] * HID + h * HDIM;
  float* o = out + (size_t)dst[e] * HID + h * HDIM;
#pragma unroll 4
  for (int c = 0; c < HDIM; ++c) atomicAdd(&o[c], v[c] * w);
}

// out_ln = LN(acc + att_r, g0, b0)
__global__ void attn_post_ln_kernel(const float* __restrict__ acc, const float* __restrict__ att_r,
                                    const float* __restrict__ g, const float* __restrict__ b,
                                    float* __restrict__ out) {
  const int row = blockIdx.x;
  __shared__ float red[256];
  const int t = threadIdx.x;
  float v[3];
#pragma unroll
  for (int i = 0; i < 3; ++i) {
    int c = t + i * 256;
    v[i] = acc[(size_t)row * HID + c] + att_r[c];
  }
  float mean = block_sum(v[0] + v[1] + v[2], red) * (1.0f / HID);
  float d0 = v[0] - mean, d1 = v[1] - mean, d2 = v[2] - mean;
  float var = block_sum(d0 * d0 + d1 * d1 + d2 * d2, red) * (1.0f / HID);
  float rstd = rsqrtf(var + LN_EPS);
#pragma unroll
  for (int i = 0; i < 3; ++i) {
    int c = t + i * 256;
    out[(size_t)row * HID + c] = (v[i] - mean) * rstd * g[c] + b[c];
  }
}

// dest = maybe_relu( LN(ln_in + relu(ff), g1, b1n) )
__global__ void resid_ln_kernel(const float* __restrict__ ln_in, const float* __restrict__ ff,
                                const float* __restrict__ g, const float* __restrict__ b,
                                float* __restrict__ out, int relu_out) {
  const int row = blockIdx.x;
  __shared__ float red[256];
  const int t = threadIdx.x;
  float v[3];
#pragma unroll
  for (int i = 0; i < 3; ++i) {
    int c = t + i * 256;
    v[i] = ln_in[(size_t)row * HID + c] + fmaxf(ff[(size_t)row * HID + c], 0.0f);
  }
  float mean = block_sum(v[0] + v[1] + v[2], red) * (1.0f / HID);
  float d0 = v[0] - mean, d1 = v[1] - mean, d2 = v[2] - mean;
  float var = block_sum(d0 * d0 + d1 * d1 + d2 * d2, red) * (1.0f / HID);
  float rstd = rsqrtf(var + LN_EPS);
#pragma unroll
  for (int i = 0; i < 3; ++i) {
    int c = t + i * 256;
    float x = (v[i] - mean) * rstd * g[c] + b[c];
    if (relu_out) x = fmaxf(x, 0.0f);
    out[(size_t)row * HID + c] = x;
  }
}

// build src/dst including per-node self edges
__global__ void build_edges_kernel(const int* __restrict__ ei, int* __restrict__ src,
                                   int* __restrict__ dst) {
  int e = blockIdx.x * blockDim.x + threadIdx.x;
  if (e >= E_TOT) return;
  if (e < E_BASE) { src[e] = ei[e]; dst[e] = ei[E_BASE + e]; }
  else            { src[e] = e - E_BASE; dst[e] = N_HEDGES + (e - E_BASE); }
}

__global__ void copy_out_kernel(const float* __restrict__ emb_s, const float* __restrict__ emb_t,
                                float* __restrict__ out) {
  long long i = (long long)blockIdx.x * blockDim.x + threadIdx.x;
  const long long a = (long long)N_NODES * HID;
  const long long tot = a + (long long)N_HEDGES * HID;
  if (i >= tot) return;
  out[i] = (i < a) ? emb_s[i] : emb_t[i - a];
}

// ---------------------------------------------------------------------------
// Host orchestration
// ---------------------------------------------------------------------------
struct AttnP {
  const float *Wk, *bk, *Wv, *bv, *att_r, *g0, *b0, *W1, *b1, *W2, *b2, *g1, *b1n;
};
static AttnP parse_attn(void* const* d_in, int base) {
  AttnP p;
  p.Wk    = (const float*)d_in[base + 0];
  p.bk    = (const float*)d_in[base + 1];
  p.Wv    = (const float*)d_in[base + 2];
  p.bv    = (const float*)d_in[base + 3];
  p.att_r = (const float*)d_in[base + 4];
  p.g0    = (const float*)d_in[base + 5];
  p.b0    = (const float*)d_in[base + 6];
  p.W1    = (const float*)d_in[base + 7];
  p.b1    = (const float*)d_in[base + 8];
  p.W2    = (const float*)d_in[base + 9];
  p.b2    = (const float*)d_in[base + 10];
  p.g1    = (const float*)d_in[base + 11];
  p.b1n   = (const float*)d_in[base + 12];
  return p;
}

extern "C" void kernel_launch(void* const* d_in, const int* in_sizes, int n_in,
                              void* d_out, int out_size, void* d_ws, size_t ws_size,
                              hipStream_t stream) {
  (void)in_sizes; (void)n_in; (void)out_size; (void)ws_size;

  const int*   x_s = (const int*)d_in[0];
  const int*   x_t = (const int*)d_in[1];
  const int*   ei  = (const int*)d_in[2];
  const float* tok = (const float*)d_in[3];
  const float* eg  = (const float*)d_in[4];
  const float* ebp = (const float*)d_in[5];

  // workspace carve-out
  size_t off = 0;
  char* base = (char*)d_ws;
  auto take = [&](size_t bytes) -> void* {
    void* p = base + off;
    off += (bytes + 255) & ~(size_t)255;
    return p;
  };
  float*          emb_s   = (float*)take((size_t)N_NODES * HID * 4);
  float*          emb_t   = (float*)take((size_t)N_TGT * HID * 4);
  float*          xK      = (float*)take((size_t)N_TGT * HID * 4);
  float*          xV      = (float*)take((size_t)N_TGT * HID * 4);
  float*          buf_out = (float*)take((size_t)N_TGT * HID * 4);
  float*          buf_ln  = (float*)take((size_t)N_TGT * HID * 4);
  float*          buf_ff  = (float*)take((size_t)N_TGT * HID * 4);
  float*          t_tem   = (float*)take((size_t)N_TGT * HID * 4);
  unsigned short* Abf     = (unsigned short*)take((size_t)N_TGT * 2 * HID * 2);
  unsigned short* Wbf     = (unsigned short*)take((size_t)2 * HID * HID * 2);
  float*          alphaB  = (float*)take((size_t)N_TGT * HEADS * 4);
  unsigned int*   amaxu   = (unsigned int*)take((size_t)N_TGT * HEADS * 4);
  float*          denom   = (float*)take((size_t)N_TGT * HEADS * 4);
  float*          exb     = (float*)take((size_t)E_TOT * HEADS * 4);
  int*            srcA    = (int*)take((size_t)E_TOT * 4);
  int*            dstA    = (int*)take((size_t)E_TOT * 4);

  auto cvt = [&](const float* in, unsigned short* out, long long rows, int cols, int ldout, int coloff) {
    long long n = rows * (long long)cols;
    cvt_kernel<<<(unsigned)((n + 255) / 256), 256, 0, stream>>>(in, out, rows, cols, ldout, coloff);
  };
  auto gemm = [&](const unsigned short* A, const unsigned short* B, const float* bias, float* C,
                  int M, int N, int K, int relu) {
    dim3 g(N / 128, (M + 127) / 128);
    gemm_bf16_wmma<<<g, 256, 0, stream>>>(A, B, bias, C, M, N, K, relu);
  };
  auto fillz = [&](void* p, long long n_u32) {
    fill_u32_kernel<<<(unsigned)((n_u32 + 255) / 256), 256, 0, stream>>>((unsigned int*)p, 0u, n_u32);
  };

  // 1) Embedding pooling + LN; emb_t = [hedge_emb ; emb_s]
  embed_pool_ln_kernel<<<N_NODES, 256, 0, stream>>>(x_s, tok, eg, ebp, emb_s);
  embed_pool_ln_kernel<<<N_HEDGES, 256, 0, stream>>>(x_t, tok, eg, ebp, emb_t);
  hipMemcpyAsync(emb_t + (size_t)N_HEDGES * HID, emb_s, (size_t)N_NODES * HID * 4,
                 hipMemcpyDeviceToDevice, stream);
  build_edges_kernel<<<(E_TOT + 255) / 256, 256, 0, stream>>>(ei, srcA, dstA);

  const unsigned eg12 = (unsigned)(((long long)E_TOT * HEADS + 255) / 256);

  auto allset = [&](const float* x, int Nsrc, const int* s_, const int* d_, int Ndst,
                    const AttnP& p, float* dest, int relu_out) {
    // projections (bf16 WMMA GEMMs)
    cvt(x, Abf, Nsrc, HID, HID, 0);
    cvt(p.Wk, Wbf, HID, HID, HID, 0);
    gemm(Abf, Wbf, p.bk, xK, Nsrc, HID, HID, 0);
    cvt(p.Wv, Wbf, HID, HID, HID, 0);
    gemm(Abf, Wbf, p.bv, xV, Nsrc, HID, HID, 0);
    // attention logits
    {
      long long n = (long long)Nsrc * HEADS;
      alpha_kernel<<<(unsigned)((n + 255) / 256), 256, 0, stream>>>(xK, p.att_r, alphaB, Nsrc);
    }
    // segment softmax + message aggregation
    fillz(amaxu, (long long)Ndst * HEADS);
    fillz(denom, (long long)Ndst * HEADS);
    fillz(buf_out, (long long)Ndst * HID);
    seg_max_kernel<<<eg12, 256, 0, stream>>>(alphaB, s_, d_, amaxu);
    seg_exp_kernel<<<eg12, 256, 0, stream>>>(alphaB, s_, d_, amaxu, denom, exb);
    seg_msg_kernel<<<eg12, 256, 0, stream>>>(exb, denom, xV, s_, d_, buf_out);
    // seed skip + LN
    attn_post_ln_kernel<<<Ndst, 256, 0, stream>>>(buf_out, p.att_r, p.g0, p.b0, buf_ln);
    // FFN
    cvt(buf_ln, Abf, Ndst, HID, HID, 0);
    cvt(p.W1, Wbf, HID, HID, HID, 0);
    gemm(Abf, Wbf, p.b1, buf_ff, Ndst, HID, HID, 1);
    cvt(buf_ff, Abf, Ndst, HID, HID, 0);
    cvt(p.W2, Wbf, HID, HID, HID, 0);
    gemm(Abf, Wbf, p.b2, buf_ff, Ndst, HID, HID, 0);
    // residual + relu + LN (+ outer relu)
    resid_ln_kernel<<<Ndst, 256, 0, stream>>>(buf_ln, buf_ff, p.g1, p.b1n, dest, relu_out);
  };

  // 2) Layers
  for (int l = 0; l < LAYERS; ++l) {
    const int lb = 6 + l * 28;
    AttnP v2e = parse_attn(d_in, lb);
    const float* Wf  = (const float*)d_in[lb + 13];
    const float* bfp = (const float*)d_in[lb + 14];
    AttnP e2v = parse_attn(d_in, lb + 15);

    // V2E: sources = nodes (emb_s), targets = 60000 hedge+self rows
    allset(emb_s, N_NODES, srcA, dstA, N_TGT, v2e, t_tem, 1);

    // emb_t = concat(emb_t, t_tem) @ Wf + bf   (K = 1536)
    cvt(emb_t, Abf, N_TGT, HID, 2 * HID, 0);
    cvt(t_tem, Abf, N_TGT, HID, 2 * HID, HID);
    cvt(Wf, Wbf, 2 * HID, HID, HID, 0);
    gemm(Abf, Wbf, bfp, emb_t, N_TGT, HID, 2 * HID, 0);

    // E2V: sources = 60000 target rows (emb_t), targets = nodes
    allset(emb_t, N_TGT, dstA, srcA, N_NODES, e2v, emb_s, 1);
  }

  // 3) Outputs: emb_s flat, then emb_t[:N_HEDGES]
  {
    long long tot = (long long)(N_NODES + N_HEDGES) * HID;
    copy_out_kernel<<<(unsigned)((tot + 255) / 256), 256, 0, stream>>>(emb_s, emb_t, (float*)d_out);
  }
}
